// Attention_49185965473896
// MI455X (gfx1250) — matrix operand order repaired and checked by reading, not compiled, not run on hardware
//
#include <hip/hip_runtime.h>
#include <cstdint>

// Problem constants (reference: B=4, T=2048, C=1024, single head, causal).
#define B_SZ   4
#define T_SEQ  2048
#define C_DIM  1024
#define M_ROWS (B_SZ * T_SEQ)   // 8192 flattened (b, t) rows

typedef __attribute__((ext_vector_type(16))) __bf16 v16bf;
typedef __attribute__((ext_vector_type(8)))  float  v8f;

union FragAB {                 // 16 bf16 = 8 VGPRs (A or B operand of 16x16x32 bf16 WMMA)
  v16bf v;
  unsigned int u[8];
  unsigned short h[16];
};
union FragC {                  // 8 f32 accumulators
  v8f v;
  float f[8];
  float4 q[2];
};

__device__ __forceinline__ unsigned short f2bf(float x) {
  unsigned int u = __float_as_uint(x);
  u += 0x7FFFu + ((u >> 16) & 1u);           // round-to-nearest-even
  return (unsigned short)(u >> 16);
}

// A-matrix (16-bit, 16x32) element-pair K offset for pair p (0..7), lane half kh.
__device__ __forceinline__ int a_off(int p, int kh) {
  return 2 * p + kh * 16;
}

__device__ __forceinline__ v8f wmma_bf16(const FragAB& a, const FragAB& b, v8f c) {
  return __builtin_amdgcn_wmma_f32_16x16x32_bf16(false, a.v, false, b.v,
                                                 (short)0, c, false, false);
}

// ---------------------------------------------------------------- converts
// Vectorized fp32 -> bf16 (4 elements per thread-iteration).
__global__ void __launch_bounds__(256)
f32_to_bf16_kernel(const float4* __restrict__ in, ushort4* __restrict__ out, int n4) {
  for (int i = blockIdx.x * blockDim.x + threadIdx.x; i < n4; i += gridDim.x * blockDim.x) {
    float4 v = in[i];
    ushort4 o;
    o.x = f2bf(v.x); o.y = f2bf(v.y); o.z = f2bf(v.z); o.w = f2bf(v.w);
    out[i] = o;
  }
}

// W [K=1024][N=1024] fp32 -> Wt [N][K] bf16, LDS-tiled 32x32 transpose
// (coalesced on both sides; +1 padding kills LDS bank conflicts).
__global__ void __launch_bounds__(256)
transpose_bf16_kernel(const float* __restrict__ in, unsigned short* __restrict__ out) {
  __shared__ float tile[32][33];
  int tx = threadIdx.x, ty = threadIdx.y;           // block (32, 8)
  int k0 = blockIdx.y * 32, n0 = blockIdx.x * 32;
#pragma unroll
  for (int i = 0; i < 32; i += 8)
    tile[ty + i][tx] = in[(size_t)(k0 + ty + i) * C_DIM + n0 + tx];
  __syncthreads();
#pragma unroll
  for (int i = 0; i < 32; i += 8)
    out[(size_t)(n0 + ty + i) * C_DIM + k0 + tx] = f2bf(tile[tx][ty + i]);
}

// ---------------------------------------------------------------- projection GEMM
// Y = A(bf16 [M][C]) @ W(bf16, stored transposed [N][K]) + bias
// One wave computes a 16(M) x 64(N) tile: 4 accumulators, A fragment reused 4x.
// MODE 0: Q  (RoPE, bf16 row-major out)
// MODE 1: K  (RoPE, bf16 row-major out)
// MODE 2: V  (no RoPE, bf16 transposed per batch: Vt[b][c][t])
// MODE 3: final projection (fp32 row-major out)
template <int MODE>
__global__ void __launch_bounds__(256)
proj_kernel(const unsigned short* __restrict__ A,
            const unsigned short* __restrict__ Wt,
            const float* __restrict__ bias,
            unsigned short* __restrict__ out16,
            float* __restrict__ out32) {
  const int NT = C_DIM / 64;                               // 16 col super-tiles
  int wid  = (int)((blockIdx.x * blockDim.x + threadIdx.x) >> 5);
  if (wid >= (M_ROWS / 16) * NT) return;
  int lane = threadIdx.x & 31;
  int tm = wid / NT, tn = wid - tm * NT;
  int m0 = tm << 4, n0 = tn << 6;
  int row = lane & 15, kh = lane >> 4;

  const unsigned short* arow = A + (size_t)(m0 + row) * C_DIM;   // A row = lane&15
  const unsigned short* brow[4];
#pragma unroll
  for (int j = 0; j < 4; ++j)
    brow[j] = Wt + (size_t)(n0 + j * 16 + row) * C_DIM;          // B col = lane&15

  FragC acc[4];
#pragma unroll
  for (int j = 0; j < 4; ++j)
#pragma unroll
    for (int r = 0; r < 8; ++r) acc[j].f[r] = 0.f;

  for (int kk = 0; kk < C_DIM; kk += 32) {
    FragAB af;
#pragma unroll
    for (int p = 0; p < 8; ++p)
      af.u[p] = *reinterpret_cast<const unsigned int*>(arow + kk + a_off(p, kh));
#pragma unroll
    for (int j = 0; j < 4; ++j) {
      FragAB bf;
#pragma unroll
      for (int p = 0; p < 8; ++p)
        bf.u[p] = *reinterpret_cast<const unsigned int*>(brow[j] + kk + kh * 16 + 2 * p);
      acc[j].v = wmma_bf16(af, bf, acc[j].v);                    // A frag reused 4x
    }
  }

#pragma unroll
  for (int j = 0; j < 4; ++j) {
    int n = n0 + j * 16 + row;            // this lane's output channel
    float bv = bias[n];
    float invf = 0.f;
    if (MODE <= 1) {
      // inv_freq = theta^(-(2i)/C), i = n>>1 ; ln(10000) = 9.2103403719...
      float i2 = (float)((n >> 1) << 1);
      invf = __expf(-(i2 / (float)C_DIM) * 9.2103403719761836f);
    }
#pragma unroll
    for (int r = 0; r < 8; ++r) {
      int m = m0 + r + 8 * kh;            // flattened (b,t) row
      float y = acc[j].f[r] + bv;
      if (MODE <= 1) {
        float prt = __shfl_xor(y, 1);     // partner channel c^1 lives in adjacent lane
        float ang = (float)(m & (T_SEQ - 1)) * invf;
        float cs = __cosf(ang), sn = __sinf(ang);
        y = (n & 1) ? (y * cs + prt * sn)   // odd channel:  x*cos + x_prev*sin
                    : (y * cs - prt * sn);  // even channel: x*cos - x_next*sin
      }
      if (MODE == 0 || MODE == 1) {
        out16[(size_t)m * C_DIM + n] = f2bf(y);
      } else if (MODE == 2) {
        int b = m >> 11, t = m & (T_SEQ - 1);
        out16[((size_t)b * C_DIM + n) * T_SEQ + t] = f2bf(y);   // Vt[b][c][t]
      } else {
        out32[(size_t)m * C_DIM + n] = y;
      }
    }
  }
}

// ---------------------------------------------------------------- flash attention
// Block = 8 waves handles one (batch, 16-query) tile. Wave w owns channels
// [128w, 128w+128): computes its partial of the 16x32 S tile, partials reduced
// via LDS; online softmax redundantly per wave; P transposed D->A layout through
// wave-private LDS; each wave accumulates O over its 8 channel tiles.
// K/V are NOT staged through LDS deliberately: no cross-wave reuse exists within
// a block, and per-batch K/V (4 MB bf16) lives in the 192 MB L2 across blocks.
__global__ void __launch_bounds__(256)
attn_kernel(const unsigned short* __restrict__ Qb,
            const unsigned short* __restrict__ Kb,
            const unsigned short* __restrict__ Vt,
            unsigned short* __restrict__ Ob) {
  __shared__ __align__(16) float sS[8][2][32][8];        // partial S: [wave][tile][lane][vgpr]
  __shared__ __align__(16) unsigned short sP[8][16][32]; // per-wave P tile, row-major

  int w = threadIdx.x >> 5, lane = threadIdx.x & 31;
  int row = lane & 15, kh = lane >> 4;
  int bq = blockIdx.x;
  int b  = bq >> 7;                         // 128 query tiles per batch
  int q0 = (bq & 127) << 4;
  const unsigned short* Q = Qb + (size_t)b * T_SEQ * C_DIM;
  const unsigned short* K = Kb + (size_t)b * T_SEQ * C_DIM;
  const unsigned short* V = Vt + (size_t)b * C_DIM * T_SEQ;
  int ch0 = w << 7;                         // this wave's channel slice

  // Q fragments for this wave's 4 K-chunks are loop-invariant: preload.
  FragAB qf[4];
#pragma unroll
  for (int c = 0; c < 4; ++c)
#pragma unroll
    for (int p = 0; p < 8; ++p)
      qf[c].u[p] = *reinterpret_cast<const unsigned int*>(
          Q + (size_t)(q0 + row) * C_DIM + ch0 + c * 32 + a_off(p, kh));

  FragC o[8];
#pragma unroll
  for (int nt = 0; nt < 8; ++nt)
#pragma unroll
    for (int r = 0; r < 8; ++r) o[nt].f[r] = 0.f;
  float mrow[8], srow[8];
#pragma unroll
  for (int r = 0; r < 8; ++r) { mrow[r] = -3.0e38f; srow[r] = 0.f; }

  int nkb = (q0 + 16 + 31) >> 5;            // causal: only key blocks <= q tile
  for (int kb = 0; kb < nkb; ++kb) {
    int k0 = kb << 5;
    // ---- partial S over this wave's 128 channels (2 tiles of 16 key cols)
    FragC s0, s1;
#pragma unroll
    for (int r = 0; r < 8; ++r) { s0.f[r] = 0.f; s1.f[r] = 0.f; }
#pragma unroll
    for (int c = 0; c < 4; ++c) {
      FragAB kf0, kf1;
      const unsigned short* k0p = K + (size_t)(k0 + row) * C_DIM + ch0 + c * 32 + kh * 16;
      const unsigned short* k1p = k0p + (size_t)16 * C_DIM;
#pragma unroll
      for (int p = 0; p < 8; ++p) {
        kf0.u[p] = *reinterpret_cast<const unsigned int*>(k0p + 2 * p);
        kf1.u[p] = *reinterpret_cast<const unsigned int*>(k1p + 2 * p);
      }
      s0.v = wmma_bf16(qf[c], kf0, s0.v);
      s1.v = wmma_bf16(qf[c], kf1, s1.v);
    }
    *reinterpret_cast<float4*>(&sS[w][0][lane][0]) = s0.q[0];
    *reinterpret_cast<float4*>(&sS[w][0][lane][4]) = s0.q[1];
    *reinterpret_cast<float4*>(&sS[w][1][lane][0]) = s1.q[0];
    *reinterpret_cast<float4*>(&sS[w][1][lane][4]) = s1.q[1];
    __syncthreads();
    // ---- reduce partials across the 8 waves (full S redundantly per wave)
    FragC S0, S1;
#pragma unroll
    for (int r = 0; r < 8; ++r) { S0.f[r] = 0.f; S1.f[r] = 0.f; }
    for (int ww = 0; ww < 8; ++ww) {
      FragC t0, t1;
      t0.q[0] = *reinterpret_cast<const float4*>(&sS[ww][0][lane][0]);
      t0.q[1] = *reinterpret_cast<const float4*>(&sS[ww][0][lane][4]);
      t1.q[0] = *reinterpret_cast<const float4*>(&sS[ww][1][lane][0]);
      t1.q[1] = *reinterpret_cast<const float4*>(&sS[ww][1][lane][4]);
      S0.v += t0.v;
      S1.v += t1.v;
    }
    __syncthreads();                        // safe to overwrite sS next block
    // ---- online softmax (lane = key col, vgpr r = query row r + 8*kh)
    int col0 = k0 + row, col1 = col0 + 16;
#pragma unroll
    for (int r = 0; r < 8; ++r) {
      int qidx = q0 + r + 8 * kh;
      float e0 = (col0 <= qidx) ? S0.f[r] * 0.03125f : -3.0e38f;  // 1/sqrt(1024)
      float e1 = (col1 <= qidx) ? S1.f[r] * 0.03125f : -3.0e38f;
      float rm = fmaxf(e0, e1);
      rm = fmaxf(rm, __shfl_xor(rm, 1));
      rm = fmaxf(rm, __shfl_xor(rm, 2));
      rm = fmaxf(rm, __shfl_xor(rm, 4));
      rm = fmaxf(rm, __shfl_xor(rm, 8));    // reduction within each 16-lane half
      float newm  = fmaxf(mrow[r], rm);
      float alpha = __expf(mrow[r] - newm);
      float p0 = __expf(e0 - newm);
      float p1 = __expf(e1 - newm);
      float rs = p0 + p1;
      rs += __shfl_xor(rs, 1);
      rs += __shfl_xor(rs, 2);
      rs += __shfl_xor(rs, 4);
      rs += __shfl_xor(rs, 8);
      srow[r] = srow[r] * alpha + rs;
      mrow[r] = newm;
#pragma unroll
      for (int nt = 0; nt < 8; ++nt) o[nt].f[r] *= alpha;
      sP[w][r + 8 * kh][row]      = f2bf(p0);      // transpose D->A layout via LDS
      sP[w][r + 8 * kh][row + 16] = f2bf(p1);
    }
    // ---- P (A layout) x V : one WMMA per 16-channel tile, K=32 key block
    FragAB pf;
#pragma unroll
    for (int p = 0; p < 8; ++p)
      pf.u[p] = *reinterpret_cast<const unsigned int*>(&sP[w][row][a_off(p, kh)]);
#pragma unroll
    for (int nt = 0; nt < 8; ++nt) {
      FragAB vf;
      const unsigned short* vp =
          V + (size_t)(ch0 + nt * 16 + row) * T_SEQ + k0 + kh * 16;
#pragma unroll
      for (int p = 0; p < 8; ++p)
        vf.u[p] = *reinterpret_cast<const unsigned int*>(vp + 2 * p);
      o[nt].v = wmma_bf16(pf, vf, o[nt].v);
    }
  }
  // ---- normalize and store O (bf16, row-major [b*T+t][c])
#pragma unroll
  for (int nt = 0; nt < 8; ++nt) {
    int cc = ch0 + nt * 16 + row;
#pragma unroll
    for (int r = 0; r < 8; ++r) {
      int m = b * T_SEQ + q0 + r + 8 * kh;
      Ob[(size_t)m * C_DIM + cc] = f2bf(o[nt].f[r] / srow[r]);
    }
  }
}

// ---------------------------------------------------------------- launch
extern "C" void kernel_launch(void* const* d_in, const int* in_sizes, int n_in,
                              void* d_out, int out_size, void* d_ws, size_t ws_size,
                              hipStream_t stream) {
  (void)in_sizes; (void)n_in; (void)out_size; (void)ws_size;
  const float* x  = (const float*)d_in[0];
  const float* Wq = (const float*)d_in[1];
  const float* bq = (const float*)d_in[2];
  const float* Wk = (const float*)d_in[3];
  const float* bk = (const float*)d_in[4];
  const float* Wv = (const float*)d_in[5];
  const float* bv = (const float*)d_in[6];
  const float* Wo = (const float*)d_in[7];
  const float* bo = (const float*)d_in[8];
  float* out = (float*)d_out;

  // Workspace carve-out (~88 MB total, 256B-aligned slabs).
  char* ws = (char*)d_ws;
  size_t off = 0;
  auto carve = [&](size_t bytes) -> void* {
    void* p = ws + off;
    off += (bytes + 255) & ~(size_t)255;
    return p;
  };
  const size_t xcb = (size_t)M_ROWS * C_DIM * sizeof(unsigned short);  // 16 MB
  const size_t wcb = (size_t)C_DIM  * C_DIM * sizeof(unsigned short);  //  2 MB
  unsigned short* xb  = (unsigned short*)carve(xcb);
  unsigned short* WqT = (unsigned short*)carve(wcb);
  unsigned short* WkT = (unsigned short*)carve(wcb);
  unsigned short* WvT = (unsigned short*)carve(wcb);
  unsigned short* WoT = (unsigned short*)carve(wcb);
  unsigned short* Qb  = (unsigned short*)carve(xcb);
  unsigned short* Kb  = (unsigned short*)carve(xcb);
  unsigned short* VtB = (unsigned short*)carve(xcb);
  unsigned short* Ob  = (unsigned short*)carve(xcb);

  // 1) precision conversion (vectorized) + LDS-tiled weight transposes
  f32_to_bf16_kernel<<<2048, 256, 0, stream>>>(
      (const float4*)x, (ushort4*)xb, (M_ROWS * C_DIM) / 4);
  dim3 tb(32, 8), tg(C_DIM / 32, C_DIM / 32);
  transpose_bf16_kernel<<<tg, tb, 0, stream>>>(Wq, WqT);
  transpose_bf16_kernel<<<tg, tb, 0, stream>>>(Wk, WkT);
  transpose_bf16_kernel<<<tg, tb, 0, stream>>>(Wv, WvT);
  transpose_bf16_kernel<<<tg, tb, 0, stream>>>(Wo, WoT);

  // 2) QKV projections (+RoPE on Q,K; V stored transposed)
  const int proj_blocks = ((M_ROWS / 16) * (C_DIM / 64)) / 8;   // 8 waves/block
  proj_kernel<0><<<proj_blocks, 256, 0, stream>>>(xb, WqT, bq, Qb,  nullptr);
  proj_kernel<1><<<proj_blocks, 256, 0, stream>>>(xb, WkT, bk, Kb,  nullptr);
  proj_kernel<2><<<proj_blocks, 256, 0, stream>>>(xb, WvT, bv, VtB, nullptr);

  // 3) causal flash attention
  attn_kernel<<<B_SZ * (T_SEQ / 16), 256, 0, stream>>>(Qb, Kb, VtB, Ob);

  // 4) output projection -> fp32
  proj_kernel<3><<<proj_blocks, 256, 0, stream>>>(Ob, WoT, bo, nullptr, out);
}
